// GridToGraphConverter_20229295964754
// MI455X (gfx1250) — compile-verified, matrix operand order checked by reading, not run on hardware
//
#include <hip/hip_runtime.h>
#include <hip/hip_bf16.h>

// Problem constants from the reference
#define BB 4
#define CC 16
#define HH 512
#define WW 512
#define HWSZ (HH * WW)          // 262144 = 2^18
#define HW_SHIFT 18
#define NODE_ELEMS (BB * HWSZ * CC)  // 16,777,216 floats

typedef __attribute__((ext_vector_type(2)))  float    v2f;
typedef __attribute__((ext_vector_type(8)))  float    v8f;
typedef __attribute__((ext_vector_type(16))) _Float16 v16h;

// ---------------------------------------------------------------------------
// Kernel 1: (B, C, H, W) -> (B*H*W, C) node features (layout transpose).
// Reads: for each channel c, lanes read consecutive pixels (coalesced 128B).
// Writes: each thread writes 4 contiguous float4 (64B contiguous per lane).
// ---------------------------------------------------------------------------
__global__ void g2g_transpose_kernel(const float* __restrict__ in,
                                     float* __restrict__ out) {
    int t = blockIdx.x * blockDim.x + threadIdx.x;   // 0 .. B*HW-1
    int b = t >> HW_SHIFT;
    int p = t & (HWSZ - 1);
    const float* src = in + (b * CC) * HWSZ + p;

    float4 r0, r1, r2, r3;
    r0.x = src[0 * HWSZ];  r0.y = src[1 * HWSZ];
    r0.z = src[2 * HWSZ];  r0.w = src[3 * HWSZ];
    r1.x = src[4 * HWSZ];  r1.y = src[5 * HWSZ];
    r1.z = src[6 * HWSZ];  r1.w = src[7 * HWSZ];
    r2.x = src[8 * HWSZ];  r2.y = src[9 * HWSZ];
    r2.z = src[10 * HWSZ]; r2.w = src[11 * HWSZ];
    r3.x = src[12 * HWSZ]; r3.y = src[13 * HWSZ];
    r3.z = src[14 * HWSZ]; r3.w = src[15 * HWSZ];

    float4* o = (float4*)(out + t * CC);
    o[0] = r0; o[1] = r1; o[2] = r2; o[3] = r3;
}

// ---------------------------------------------------------------------------
// Kernel 2: one wave32 per 16 edges.
// Lane L: edge m = L&15, batch pair = (L>>4)*2 + {0,1}.
// Partial abs-diff sums reduced across the 4 batch-partials per edge with
// V_WMMA_F32_16X16X4_F32 (A = 16x4 partials, B = ones) -> D[m][*] = total.
// ---------------------------------------------------------------------------
__global__ void g2g_edge_kernel(const int* __restrict__ ei,
                                const float* __restrict__ nf,
                                float* __restrict__ out_ei,
                                float* __restrict__ out_attr,
                                int E) {
    int tid  = blockIdx.x * blockDim.x + threadIdx.x;
    int wave = tid >> 5;
    int lane = threadIdx.x & 31;

    int base = wave * 16;          // first edge of this wave
    if (base >= E) return;         // wave-uniform: EXEC stays all-1s

    int m  = lane & 15;
    int hf = lane >> 4;            // 0: batches {0,1}, 1: batches {2,3}
    int e  = base + m;
    int ec = (e < E) ? e : (E - 1);  // branchless clamp for tail

    int src = ei[ec];
    int dst = ei[E + ec];

    // two batch partial sums of |src - dst| over 16 channels each
    float part[2];
    #pragma unroll
    for (int bb = 0; bb < 2; ++bb) {
        int b = hf * 2 + bb;
        const float4* ps = (const float4*)(nf + (((b << HW_SHIFT) + src) << 4));
        const float4* pd = (const float4*)(nf + (((b << HW_SHIFT) + dst) << 4));
        float acc = 0.0f;
        #pragma unroll
        for (int i = 0; i < 4; ++i) {
            float4 a = ps[i];
            float4 c = pd[i];
            acc += fabsf(a.x - c.x) + fabsf(a.y - c.y) +
                   fabsf(a.z - c.z) + fabsf(a.w - c.w);
        }
        part[bb] = acc;
    }
    // zero out tail edges so they don't pollute (no branch: EXEC stays full)
    float p0 = (e < E) ? part[0] : 0.0f;
    float p1 = (e < E) ? part[1] : 0.0f;

    v8f cz = {};
    v8f d;
#if __has_builtin(__builtin_amdgcn_wmma_f32_16x16x4_f32)
    // A 16x4 f32 layout: lane<16 -> (M=lane, K0,K1); lane>=16 -> (M, K2,K3)
    v2f a; a.x = p0; a.y = p1;
    v2f bones; bones.x = 1.0f; bones.y = 1.0f;
    d = __builtin_amdgcn_wmma_f32_16x16x4_f32(false, a, false, bones,
                                              (short)0, cz, false, false);
#else
    // Fallback: codegen-confirmed f16 WMMA. Put the 2 partials in this lane's
    // first two K-slots (K0,K1 for lanes<16; K8,K9 for lanes>=16), B = ones.
    v16h a = (v16h)(_Float16)0.0f;
    a[0] = (_Float16)p0;
    a[1] = (_Float16)p1;
    v16h bones = (v16h)(_Float16)1.0f;
    d = __builtin_amdgcn_wmma_f32_16x16x32_f16(false, a, false, bones,
                                               (short)0, cz, false, false);
#endif

    // D (16x16 f32): VGPR r, lanes 0-15 hold M=r, lanes 16-31 hold M=r+8;
    // value replicated across all N columns.
    // feature_diff writers: lanes 0-7 -> edges 0-7 (d[lane]);
    //                       lanes 16-23 -> edges 8-15 (d[lane-16]).
    int r = lane & 7;
    float total = d[r];

    if (lane < 8) {
        int ee = base + lane;
        if (ee < E) out_attr[ee * 2 + 1] = total * (1.0f / 64.0f);
    } else if (lane >= 16 && lane < 24) {
        int ee = base + 8 + (lane - 16);
        if (ee < E) out_attr[ee * 2 + 1] = total * (1.0f / 64.0f);
    }

    // spatial_dist + edge_index copy: lanes 0-15 own edge (base+m)'s src/dst
    if (lane < 16 && e < E) {
        int sh = src >> 9, sw = src & (WW - 1);
        int dh = dst >> 9, dw = dst & (WW - 1);
        int ddh = sh - dh, ddw = sw - dw;
        out_attr[e * 2 + 0] = sqrtf((float)(ddh * ddh + ddw * ddw));
        out_ei[e]     = (float)src;
        out_ei[E + e] = (float)dst;
    }
}

extern "C" void kernel_launch(void* const* d_in, const int* in_sizes, int n_in,
                              void* d_out, int out_size, void* d_ws, size_t ws_size,
                              hipStream_t stream) {
    const float* grid = (const float*)d_in[0];
    const int*   ei   = (const int*)d_in[1];
    int E = in_sizes[1] / 2;

    float* out      = (float*)d_out;
    float* node     = out;                       // NODE_ELEMS floats
    float* out_ei   = out + NODE_ELEMS;          // 2*E floats
    float* out_attr = out_ei + (size_t)2 * E;    // 2*E floats

    // Kernel 1: transpose into node-features region
    {
        int threads = 256;
        int blocks  = (BB * HWSZ) / threads;     // 4096
        g2g_transpose_kernel<<<blocks, threads, 0, stream>>>(grid, node);
    }
    // Kernel 2: edge attributes + edge_index copy (reads node region)
    {
        int waves   = (E + 15) / 16;
        int threads = 256;                       // 8 waves per block
        int blocks  = (waves + 7) / 8;
        g2g_edge_kernel<<<blocks, threads, 0, stream>>>(ei, node, out_ei,
                                                        out_attr, E);
    }
}
